// MambaHead_37641093382897
// MI455X (gfx1250) — compile-verified, matrix operand order checked by reading, not run on hardware
//
#include <hip/hip_runtime.h>
#include <hip/hip_bf16.h>

// ---- problem constants (from reference) ----
#define BSZ     8
#define SEQ     2048
#define DMODEL  768
#define DINNER  1536
#define DSTATE  16
#define DTRANK  48          // DT_RANK
#define XDBLC   80          // DT_RANK + 2*D_STATE
#define NTOK    (BSZ*SEQ)   // 16384
#define LABLEN  3
#define LABDIM  2

typedef __attribute__((ext_vector_type(8)))  __bf16 v8bf;
typedef __attribute__((ext_vector_type(16))) __bf16 v16bf;
typedef __attribute__((ext_vector_type(8)))  float  v8f;

// ---- gfx1250 async load-to-LDS availability probes ----
#if defined(__has_builtin)
# if __has_builtin(__builtin_amdgcn_global_load_async_to_lds_b128)
#  define HAVE_ASYNC_LDS 1
# endif
# if __has_builtin(__builtin_amdgcn_s_wait_asynccnt)
#  define HAVE_WAIT_ASYNC 1
# endif
#endif

#if defined(HAVE_ASYNC_LDS)
// builtin signature (from hipcc diagnostic): (v4i AS1*, v4i AS3*, imm offset, imm cpol)
typedef int v4i_t __attribute__((vector_size(16)));
typedef __attribute__((address_space(1))) v4i_t* g_v4i_p;
typedef __attribute__((address_space(3))) v4i_t* l_v4i_p;
#endif

// 16-byte global -> LDS copy. Async (ASYNCcnt-tracked) when available.
__device__ __forceinline__ void cp16_g2l(const __bf16* g, __bf16* l) {
#if defined(HAVE_ASYNC_LDS)
    g_v4i_p gp = (g_v4i_p)(uintptr_t)g;
    // generic LDS address: low 32 bits are the LDS byte offset
    l_v4i_p lp = (l_v4i_p)(uintptr_t)(uint32_t)(uintptr_t)l;
    __builtin_amdgcn_global_load_async_to_lds_b128(gp, lp, 0, 0);
#else
    *(v8bf*)l = *(const v8bf*)g;
#endif
}

__device__ __forceinline__ void wait_async_all() {
#if defined(HAVE_ASYNC_LDS)
# if defined(HAVE_WAIT_ASYNC)
    __builtin_amdgcn_s_wait_asynccnt(0);
# else
    asm volatile("s_wait_asynccnt 0" ::: "memory");
# endif
#endif
}

// ---------------- WMMA helpers (CDNA5, wave32) ----------------
// Fragment layout per CDNA5 ISA 7.12.2 (16-bit A 16x32):
// lanes 0-15 hold K 0..7 and 16..23 (rows M=lane); lanes 16-31 hold K 8..15 and 24..31.
// B (32x16) loaded from a pre-transposed N-major buffer uses the identical pattern.
__device__ __forceinline__ v16bf ld_frag(const __bf16* p) {
    v8bf lo = *(const v8bf*)p;
    v8bf hi = *(const v8bf*)(p + 16);
    v16bf r;
#pragma unroll
    for (int i = 0; i < 8; ++i) { r[i] = lo[i]; r[i + 8] = hi[i]; }
    return r;
}

__device__ __forceinline__ v8f wmma_bf16(v16bf a, v16bf b, v8f c) {
    return __builtin_amdgcn_wmma_f32_16x16x32_bf16(false, a, false, b, (short)0, c, false, false);
}

// C/D 16x16 f32 layout: lane -> N (lane&15), vgpr r -> M = r + (lane>=16 ? 8 : 0)
__device__ __forceinline__ void store_tile(float* C, int ldc, int rowBase, int col, const v8f& acc) {
#pragma unroll
    for (int r = 0; r < 8; ++r) C[(size_t)(rowBase + r) * ldc + col] = acc[r];
}

// ---------------- weight repack kernels (tiny) ----------------
__global__ void k_pack_w_inT(const float* __restrict__ w_in, __bf16* __restrict__ w_inT) {
    int idx = blockIdx.x * blockDim.x + threadIdx.x;          // n*768 + k
    int n = idx / DMODEL, k = idx - n * DMODEL;
    w_inT[idx] = (__bf16)w_in[(size_t)k * (2 * DINNER) + n];
}
__global__ void k_pack_w_xT(const float* __restrict__ w_x, __bf16* __restrict__ w_xT) {
    int idx = blockIdx.x * blockDim.x + threadIdx.x;          // n*1536 + k
    int n = idx / DINNER, k = idx - n * DINNER;
    w_xT[idx] = (__bf16)w_x[(size_t)k * XDBLC + n];
}
__global__ void k_pack_w_dtT(const float* __restrict__ w_dt, __bf16* __restrict__ w_dtT) {
    int idx = blockIdx.x * blockDim.x + threadIdx.x;          // n*64 + k
    int n = idx >> 6, k = idx & 63;
    w_dtT[idx] = (k < DTRANK) ? (__bf16)w_dt[(size_t)k * DINNER + n] : (__bf16)0.0f;
}
__global__ void k_pack_dt(const float* __restrict__ xdbl, __bf16* __restrict__ dtb) {
    int idx = blockIdx.x * blockDim.x + threadIdx.x;          // row*64 + k
    int r = idx >> 6, k = idx & 63;
    dtb[idx] = (k < DTRANK) ? (__bf16)xdbl[(size_t)r * XDBLC + k] : (__bf16)0.0f;
}

// ---------------- rmsnorm(x) -> bf16 ----------------
__global__ void k_rmsnorm_bf16(const float* __restrict__ x, const float* __restrict__ w,
                               __bf16* __restrict__ h) {
    __shared__ float red[256];
    int g = blockIdx.x, tid = threadIdx.x;
    const float* row = x + (size_t)g * DMODEL;
    float v[3]; float s = 0.f;
#pragma unroll
    for (int k = 0; k < 3; ++k) { v[k] = row[tid + k * 256]; s += v[k] * v[k]; }
    red[tid] = s; __syncthreads();
    for (int st = 128; st > 0; st >>= 1) { if (tid < st) red[tid] += red[tid + st]; __syncthreads(); }
    float rs = rsqrtf(red[0] / DMODEL + 1e-5f);
#pragma unroll
    for (int k = 0; k < 3; ++k) {
        int j = tid + k * 256;
        h[(size_t)g * DMODEL + j] = (__bf16)(v[k] * rs * w[j]);
    }
}

// ---------------- WMMA GEMM, async-LDS staged B: C[MxN] = A[MxK] * BT[NxK]^T -------
// block = 256 threads = 8 waves as 4(M) x 2(N); wave tile 32x64; block tile 128x128.
// B panel (128 x 32 bf16 = 8KB) double-buffered in LDS via global_load_async_to_lds_b128
// (ASYNCcnt + barrier sync); A streamed straight to registers with global_prefetch ahead.
// grid (N/128, M/128).  K % 32 == 0.
__global__ void __launch_bounds__(256)
k_gemm_bf16_nt(const __bf16* __restrict__ A, const __bf16* __restrict__ BT,
               float* __restrict__ C, int K, int N) {
    __shared__ __align__(16) __bf16 Blds[2][128 * 32];

    int lane = threadIdx.x & 31, wave = threadIdx.x >> 5;
    int wm = wave & 3;                    // 0..3 -> M
    int wn = wave >> 2;                   // 0..1 -> N
    int mBase = blockIdx.y * 128 + wm * 32;
    int nBlk  = blockIdx.x * 128;
    int lr = lane & 15;
    int koff = (lane >> 4) << 3;          // 0 or 8

    const __bf16* a0 = A + (size_t)(mBase + lr) * K;
    const __bf16* a1 = a0 + (size_t)16 * K;

    // each thread copies 2 x 16B chunks of the B panel per k-step
    int c0 = threadIdx.x * 2;
    int row0 = c0 >> 2,      ko0 = (c0 & 3) * 8;
    int row1 = (c0 + 1) >> 2, ko1 = ((c0 + 1) & 3) * 8;

    v8f acc[2][4];
#pragma unroll
    for (int i = 0; i < 2; ++i)
#pragma unroll
        for (int j = 0; j < 4; ++j) acc[i][j] = (v8f){};

    // prologue: stage k-step 0
    cp16_g2l(BT + (size_t)(nBlk + row0) * K + ko0, &Blds[0][row0 * 32 + ko0]);
    cp16_g2l(BT + (size_t)(nBlk + row1) * K + ko1, &Blds[0][row1 * 32 + ko1]);

    int nsteps = K >> 5;
    for (int s = 0; s < nsteps; ++s) {
        wait_async_all();
        __syncthreads();                              // panel s ready; panel s-1 reads done
        if (s + 1 < nsteps) {
            int kk1 = (s + 1) << 5;
            int nb = (s + 1) & 1;
            cp16_g2l(BT + (size_t)(nBlk + row0) * K + kk1 + ko0, &Blds[nb][row0 * 32 + ko0]);
            cp16_g2l(BT + (size_t)(nBlk + row1) * K + kk1 + ko1, &Blds[nb][row1 * 32 + ko1]);
        }
        int kk = s << 5;
        __builtin_prefetch(a0 + kk + 256, 0, 0);      // stream A into GL2
        v16bf fa0 = ld_frag(a0 + kk + koff);
        v16bf fa1 = ld_frag(a1 + kk + koff);
        const __bf16* bp = &Blds[s & 1][(wn * 64 + lr) * 32 + koff];
#pragma unroll
        for (int j = 0; j < 4; ++j) {
            v16bf fb = ld_frag(bp + j * (16 * 32));
            acc[0][j] = wmma_bf16(fa0, fb, acc[0][j]);
            acc[1][j] = wmma_bf16(fa1, fb, acc[1][j]);
        }
    }

    int rb = mBase + ((lane >> 4) << 3);
#pragma unroll
    for (int i = 0; i < 2; ++i)
#pragma unroll
        for (int j = 0; j < 4; ++j)
            store_tile(C, N, rb + i * 16, nBlk + wn * 64 + j * 16 + lr, acc[i][j]);
}

// ---------------- xdbl GEMM: [16384x80] = xc_bf16 * w_xT[80x1536]^T ----------------
__global__ void k_gemm_xdbl(const __bf16* __restrict__ A, const __bf16* __restrict__ BT,
                            float* __restrict__ C) {
    int lane = threadIdx.x & 31, wave = threadIdx.x >> 5;
    int mBase = blockIdx.x * 128 + wave * 16;
    int lr = lane & 15;
    int koff = (lane >> 4) << 3;
    const __bf16* a = A + (size_t)(mBase + lr) * DINNER;
    v8f acc[5] = {};
    for (int kk = 0; kk < DINNER; kk += 32) {
        v16bf fa = ld_frag(a + kk + koff);
#pragma unroll
        for (int nt = 0; nt < 5; ++nt) {
            v16bf fb = ld_frag(BT + (size_t)(nt * 16 + lr) * DINNER + kk + koff);
            acc[nt] = wmma_bf16(fa, fb, acc[nt]);
        }
    }
    int rb = mBase + ((lane >> 4) << 3);
#pragma unroll
    for (int nt = 0; nt < 5; ++nt) store_tile(C, XDBLC, rb, nt * 16 + lr, acc[nt]);
}

// ---------------- causal depthwise conv (K=4) + SiLU; emits f32 + bf16 ----------------
__global__ void k_conv_silu(const float* __restrict__ xi, const float* __restrict__ w_conv,
                            const float* __restrict__ b_conv,
                            float* __restrict__ xc, __bf16* __restrict__ xcb) {
    int idx = blockIdx.x * blockDim.x + threadIdx.x;         // row*1536 + d
    int row = idx / DINNER, d = idx - row * DINNER;
    int t = row & (SEQ - 1);
    float acc = b_conv[d];
#pragma unroll
    for (int j = 0; j < 4; ++j) {
        int tt = t - 3 + j;
        if (tt >= 0) acc += xi[(size_t)(row - 3 + j) * DINNER + d] * w_conv[d * 4 + j];
    }
    float s = acc / (1.f + __expf(-acc));
    xc[idx] = s;
    xcb[idx] = (__bf16)s;
}

// ---------------- softplus(raw + b_dt[col]) in place ----------------
__global__ void k_softplus_bias(float* __restrict__ delta, const float* __restrict__ b_dt) {
    int idx = blockIdx.x * blockDim.x + threadIdx.x;
    int col = idx % DINNER;
    float v = delta[idx] + b_dt[col];
    delta[idx] = (v > 20.f) ? v : log1pf(__expf(v));
}

// ---------------- selective scan: one thread per (b,d), 16-state in regs -----------
__global__ void k_scan(const float* __restrict__ xc, const float* __restrict__ delta,
                       const float* __restrict__ xdbl, const float* __restrict__ A_log,
                       const float* __restrict__ D_skip, float* __restrict__ y24) {
    int tid = blockIdx.x * blockDim.x + threadIdx.x;         // 0..12287
    int b = tid / DINNER, d = tid - b * DINNER;
    float a[DSTATE];
#pragma unroll
    for (int n = 0; n < DSTATE; ++n) a[n] = -__expf(A_log[d * DSTATE + n]);
    float h[DSTATE];
#pragma unroll
    for (int n = 0; n < DSTATE; ++n) h[n] = 0.f;
    const float* dp = delta + (size_t)b * SEQ * DINNER + d;
    const float* up = xc    + (size_t)b * SEQ * DINNER + d;
    const float  Dd = D_skip[d];
    for (int t = 0; t < SEQ; ++t) {
        float dlt = dp[(size_t)t * DINNER];
        float u   = up[(size_t)t * DINNER];
        float du  = dlt * u;
        const float* brow = xdbl + ((size_t)b * SEQ + t) * XDBLC + DTRANK;   // B[16], then C[16]
#pragma unroll
        for (int n = 0; n < DSTATE; ++n)
            h[n] = h[n] * __expf(dlt * a[n]) + du * brow[n];
        if (t >= SEQ - LABLEN) {
            const float* crow = brow + DSTATE;
            float y = u * Dd;
#pragma unroll
            for (int n = 0; n < DSTATE; ++n) y += h[n] * crow[n];
            y24[(size_t)(b * LABLEN + (t - (SEQ - LABLEN))) * DINNER + d] = y;
        }
    }
}

// ---------------- z-gate for 24 rows: y *= silu(h_row @ w_in[:,1536+c]) -----------
__global__ void k_zgate(const __bf16* __restrict__ h, const float* __restrict__ w_in,
                        float* __restrict__ y24) {
    int i = blockIdx.x;                                      // 0..23
    int b = i / LABLEN, r = i - b * LABLEN, t = SEQ - LABLEN + r;
    const __bf16* hrow = h + (size_t)(b * SEQ + t) * DMODEL;
    for (int c = threadIdx.x; c < DINNER; c += blockDim.x) {
        float z = 0.f;
        for (int k = 0; k < DMODEL; ++k)
            z += (float)hrow[k] * w_in[(size_t)k * (2 * DINNER) + DINNER + c];
        float sg = z / (1.f + __expf(-z));
        y24[(size_t)i * DINNER + c] *= sg;
    }
}

// ---------------- out24 = y24 @ w_out + residual ----------------
__global__ void k_outproj(const float* __restrict__ y24, const float* __restrict__ w_out,
                          const float* __restrict__ x, float* __restrict__ out24) {
    int idx = blockIdx.x * blockDim.x + threadIdx.x;         // i*768 + j
    int i = idx / DMODEL, j = idx - i * DMODEL;
    int b = i / LABLEN, r = i - b * LABLEN, t = SEQ - LABLEN + r;
    float acc = x[((size_t)b * SEQ + t) * DMODEL + j];
    const float* yrow = y24 + (size_t)i * DINNER;
    for (int c = 0; c < DINNER; ++c) acc += yrow[c] * w_out[(size_t)c * DMODEL + j];
    out24[idx] = acc;
}

// ---------------- final rmsnorm + head over 24 rows ----------------
__global__ void k_norm_head(const float* __restrict__ out24, const float* __restrict__ normf_w,
                            const float* __restrict__ w_head, const float* __restrict__ b_head,
                            float* __restrict__ out) {
    __shared__ float red[256];
    int i = blockIdx.x, tid = threadIdx.x;
    const float* row = out24 + (size_t)i * DMODEL;
    float v[3]; float s = 0.f;
#pragma unroll
    for (int k = 0; k < 3; ++k) { v[k] = row[tid + k * 256]; s += v[k] * v[k]; }
    red[tid] = s; __syncthreads();
    for (int st = 128; st > 0; st >>= 1) { if (tid < st) red[tid] += red[tid + st]; __syncthreads(); }
    float rs = rsqrtf(red[0] / DMODEL + 1e-5f);
    __syncthreads();
    float p0 = 0.f, p1 = 0.f;
#pragma unroll
    for (int k = 0; k < 3; ++k) {
        int j = tid + k * 256;
        float val = v[k] * rs * normf_w[j];
        p0 += val * w_head[j * LABDIM + 0];
        p1 += val * w_head[j * LABDIM + 1];
    }
    red[tid] = p0; __syncthreads();
    for (int st = 128; st > 0; st >>= 1) { if (tid < st) red[tid] += red[tid + st]; __syncthreads(); }
    float r0 = red[0]; __syncthreads();
    red[tid] = p1; __syncthreads();
    for (int st = 128; st > 0; st >>= 1) { if (tid < st) red[tid] += red[tid + st]; __syncthreads(); }
    if (tid == 0) {
        out[i * LABDIM + 0] = r0 + b_head[0];
        out[i * LABDIM + 1] = red[0] + b_head[1];
    }
}

// ---------------- host driver ----------------
static inline void* wsTake(char*& p, size_t bytes) {
    void* r = (void*)p;
    p += (bytes + 255) & ~(size_t)255;
    return r;
}

extern "C" void kernel_launch(void* const* d_in, const int* in_sizes, int n_in,
                              void* d_out, int out_size, void* d_ws, size_t ws_size,
                              hipStream_t stream) {
    (void)in_sizes; (void)n_in; (void)out_size; (void)ws_size;
    const float* x      = (const float*)d_in[0];
    const float* w_in   = (const float*)d_in[1];
    const float* w_conv = (const float*)d_in[2];
    const float* b_conv = (const float*)d_in[3];
    const float* w_x    = (const float*)d_in[4];
    const float* w_dt   = (const float*)d_in[5];
    const float* b_dt   = (const float*)d_in[6];
    const float* A_log  = (const float*)d_in[7];
    const float* D_skip = (const float*)d_in[8];
    const float* w_out  = (const float*)d_in[9];
    const float* norm_w = (const float*)d_in[10];
    const float* normf_w= (const float*)d_in[11];
    const float* w_head = (const float*)d_in[12];
    const float* b_head = (const float*)d_in[13];
    float* out = (float*)d_out;

    char* wp = (char*)d_ws;
    __bf16* h_bf16 = (__bf16*)wsTake(wp, (size_t)NTOK * DMODEL * 2);
    __bf16* w_inT  = (__bf16*)wsTake(wp, (size_t)DINNER * DMODEL * 2);
    __bf16* w_xT   = (__bf16*)wsTake(wp, (size_t)XDBLC * DINNER * 2);
    __bf16* w_dtT  = (__bf16*)wsTake(wp, (size_t)DINNER * 64 * 2);
    float*  xi     = (float*) wsTake(wp, (size_t)NTOK * DINNER * 4);
    float*  xc     = (float*) wsTake(wp, (size_t)NTOK * DINNER * 4);
    __bf16* xcb    = (__bf16*)wsTake(wp, (size_t)NTOK * DINNER * 2);
    float*  xdbl   = (float*) wsTake(wp, (size_t)NTOK * XDBLC * 4);
    __bf16* dtb    = (__bf16*)wsTake(wp, (size_t)NTOK * 64 * 2);
    float*  delta  = (float*) wsTake(wp, (size_t)NTOK * DINNER * 4);
    float*  y24    = (float*) wsTake(wp, (size_t)BSZ * LABLEN * DINNER * 4);
    float*  out24  = (float*) wsTake(wp, (size_t)BSZ * LABLEN * DMODEL * 4);

    // weight repacks (bf16, N-major for WMMA B operand)
    k_pack_w_inT<<<(DINNER * DMODEL) / 256, 256, 0, stream>>>(w_in, w_inT);
    k_pack_w_xT <<<(XDBLC * DINNER) / 256, 256, 0, stream>>>(w_x, w_xT);
    k_pack_w_dtT<<<(DINNER * 64) / 256, 256, 0, stream>>>(w_dt, w_dtT);

    // rmsnorm -> bf16 activations
    k_rmsnorm_bf16<<<NTOK, 256, 0, stream>>>(x, norm_w, h_bf16);

    // xi = h @ w_in[:, :1536]   (M=16384, N=1536, K=768)
    k_gemm_bf16_nt<<<dim3(DINNER / 128, NTOK / 128), 256, 0, stream>>>(h_bf16, w_inT, xi, DMODEL, DINNER);

    // conv + silu
    k_conv_silu<<<(NTOK * DINNER) / 256, 256, 0, stream>>>(xi, w_conv, b_conv, xc, xcb);

    // xdbl = xc @ w_x          (M=16384, N=80, K=1536)
    k_gemm_xdbl<<<NTOK / 128, 256, 0, stream>>>(xcb, w_xT, xdbl);

    // dt -> bf16 (K padded to 64), delta = softplus(dt @ w_dt + b_dt)
    k_pack_dt<<<(NTOK * 64) / 256, 256, 0, stream>>>(xdbl, dtb);
    k_gemm_bf16_nt<<<dim3(DINNER / 128, NTOK / 128), 256, 0, stream>>>(dtb, w_dtT, delta, 64, DINNER);
    k_softplus_bias<<<(NTOK * DINNER) / 256, 256, 0, stream>>>(delta, b_dt);

    // selective scan (emit last 3 timesteps only)
    k_scan<<<(BSZ * DINNER) / 256, 256, 0, stream>>>(xc, delta, xdbl, A_log, D_skip, y24);

    // gate (z computed only for 24 rows), out-projection + residual, final norm + head
    k_zgate<<<BSZ * LABLEN, 256, 0, stream>>>(h_bf16, w_in, y24);
    k_outproj<<<(BSZ * LABLEN * DMODEL) / 256, 256, 0, stream>>>(y24, w_out, x, out24);
    k_norm_head<<<BSZ * LABLEN, 256, 0, stream>>>(out24, normf_w, w_head, b_head, out);
}
